// RepresentationSummarizerSingleInstance_30073361006607
// MI455X (gfx1250) — matrix operand order repaired and checked by reading, not compiled, not run on hardware
//
#include <hip/hip_runtime.h>
#include <hip/hip_bf16.h>

#define NRES   700
#define BATCH  128
#define DIN    1024
#define HID    1024
#define LDS_STRIDE 1040          // ushorts per row: 1024 + 16 pad (2080 B, avoids bank conflicts)

typedef __attribute__((ext_vector_type(16))) __bf16 v16bf;
typedef __attribute__((ext_vector_type(8)))  __bf16 v8bf;
typedef __attribute__((ext_vector_type(8)))  float  v8f;

__device__ __forceinline__ unsigned short f32_to_bf16(float f) {
    unsigned u = __float_as_uint(f);
    u += 0x7FFFu + ((u >> 16) & 1u);   // round-to-nearest-even
    return (unsigned short)(u >> 16);
}

// ---------------------------------------------------------------------------
// Kernel 1: pack W1 (fp32 [D,H] row-major) into per-WMMA-fragment bf16 layout.
// tile id = ht*32 + kt  (64 h-tiles x 32 k-tiles), 512 bf16 per tile.
// Within a tile: lane l (0..31) holds B[k = kt*32 + 8*(l/16) + {0..7,16..23}, n = ht*16 + l%16],
// contiguous 32 bytes -> one lane = two global_load_b128 in the GEMM.
// ---------------------------------------------------------------------------
__global__ void pack_w1_bf16(const float* __restrict__ W1,
                             unsigned short* __restrict__ wb) {
    int p    = blockIdx.x * 256 + threadIdx.x;   // 0 .. 1M-1
    int tile = p >> 9;
    int l    = (p >> 4) & 31;
    int n    = l & 15;
    int ht   = tile >> 5;
    int kt   = tile & 31;
    int k    = kt * 32 + ((l >> 4) << 3) + (p & 7) + (((p >> 3) & 1) << 4);
    wb[p] = f32_to_bf16(W1[k * HID + ht * 16 + n]);
}

// ---------------------------------------------------------------------------
// Kernel 2: fused  y = relu(X @ W1 + b1) @ W2 + b2  for one residue slice.
// Block = residue r: stage X[r] (128x1024) to LDS as bf16 once; each of the
// 8 waves owns 16 rows and sweeps all 1024 hidden units, 4 h-tiles per pass
// (4 independent WMMA accumulation chains for XDL pipelining).
// ---------------------------------------------------------------------------
__global__ void __launch_bounds__(256) fused_mlp(
    const float* __restrict__ X,              // [700,128,1024] f32
    const unsigned short* __restrict__ WB,    // packed bf16 W1 fragments
    const float* __restrict__ b1,             // [1024]
    const float* __restrict__ W2,             // [1024]
    const float* __restrict__ b2,             // [1]
    float* __restrict__ out)                  // [128,700]
{
    extern __shared__ char smem_raw[];
    unsigned short* sx = (unsigned short*)smem_raw;

    const int r   = blockIdx.x;
    const int tid = threadIdx.x;

    // ---- Stage X[r] fp32 -> bf16 LDS (coalesced float4 reads) ----
    const float* xr = X + (size_t)r * (BATCH * DIN);
    for (int idx = tid * 4; idx < BATCH * DIN; idx += 256 * 4) {
        const float4 v = *(const float4*)(xr + idx);
        int row = idx >> 10;
        int col = idx & 1023;
        uint2 pk;
        pk.x = (unsigned)f32_to_bf16(v.x) | ((unsigned)f32_to_bf16(v.y) << 16);
        pk.y = (unsigned)f32_to_bf16(v.z) | ((unsigned)f32_to_bf16(v.w) << 16);
        *(uint2*)(sx + row * LDS_STRIDE + col) = pk;
    }
    __syncthreads();

    const int lane = tid & 31;
    const int wave = tid >> 5;
    // A-fragment base per ISA 16-bit A layout:
    //   lanes 0-15: row M=lane,    K chunks {0..7, 16..23}
    //   lanes16-31: row M=lane-16, K chunks {8..15, 24..31}
    const int mrow = wave * 16 + (lane & 15);
    const int kb   = (lane >> 4) << 3;                 // 0 or 8
    const unsigned short* arow = sx + mrow * LDS_STRIDE + kb;

    float yacc[8] = {0.f,0.f,0.f,0.f,0.f,0.f,0.f,0.f};

    for (int hq = 0; hq < HID / 64; ++hq) {            // quads of 16-wide h-tiles
        const int ht0 = hq * 4, ht1 = ht0 + 1, ht2 = ht0 + 2, ht3 = ht0 + 3;
        const v16bf* bp0 = ((const v16bf*)WB) + (size_t)(ht0 * 32) * 32 + lane;
        const v16bf* bp1 = ((const v16bf*)WB) + (size_t)(ht1 * 32) * 32 + lane;
        const v16bf* bp2 = ((const v16bf*)WB) + (size_t)(ht2 * 32) * 32 + lane;
        const v16bf* bp3 = ((const v16bf*)WB) + (size_t)(ht3 * 32) * 32 + lane;

        v8f c0 = {0.f,0.f,0.f,0.f,0.f,0.f,0.f,0.f};
        v8f c1 = {0.f,0.f,0.f,0.f,0.f,0.f,0.f,0.f};
        v8f c2 = {0.f,0.f,0.f,0.f,0.f,0.f,0.f,0.f};
        v8f c3 = {0.f,0.f,0.f,0.f,0.f,0.f,0.f,0.f};

        #pragma unroll 2
        for (int kt = 0; kt < 32; ++kt) {
            const unsigned short* ap = arow + kt * 32;   // K step of 32 elems
            v8bf alo = *(const v8bf*)(ap);               // K kb..kb+7
            v8bf ahi = *(const v8bf*)(ap + 16);          // K kb+16..kb+23
            v16bf a;
            #pragma unroll
            for (int i = 0; i < 8; ++i) { a[i] = alo[i]; a[8 + i] = ahi[i]; }

            v16bf bf0 = bp0[kt * 32];
            v16bf bf1 = bp1[kt * 32];
            v16bf bf2 = bp2[kt * 32];
            v16bf bf3 = bp3[kt * 32];

            c0 = __builtin_amdgcn_wmma_f32_16x16x32_bf16(false, a, false, bf0,
                                                         (short)0, c0, false, false);
            c1 = __builtin_amdgcn_wmma_f32_16x16x32_bf16(false, a, false, bf1,
                                                         (short)0, c1, false, false);
            c2 = __builtin_amdgcn_wmma_f32_16x16x32_bf16(false, a, false, bf2,
                                                         (short)0, c2, false, false);
            c3 = __builtin_amdgcn_wmma_f32_16x16x32_bf16(false, a, false, bf3,
                                                         (short)0, c3, false, false);
        }

        // Epilogue: +b1, relu, *W2, accumulate per-row scalar.
        // C layout: VGPR j, lanes0-15 -> (M=j, N=lane); lanes16-31 -> (M=j+8).
        const int n = lane & 15;
        const float w20 = W2[ht0 * 16 + n], w21 = W2[ht1 * 16 + n];
        const float w22 = W2[ht2 * 16 + n], w23 = W2[ht3 * 16 + n];
        const float bb0 = b1[ht0 * 16 + n], bb1 = b1[ht1 * 16 + n];
        const float bb2 = b1[ht2 * 16 + n], bb3 = b1[ht3 * 16 + n];
        #pragma unroll
        for (int j = 0; j < 8; ++j) {
            yacc[j] += fmaxf(c0[j] + bb0, 0.0f) * w20;
            yacc[j] += fmaxf(c1[j] + bb1, 0.0f) * w21;
            yacc[j] += fmaxf(c2[j] + bb2, 0.0f) * w22;
            yacc[j] += fmaxf(c3[j] + bb3, 0.0f) * w23;
        }
    }

    // Reduce over the 16 N-lanes of each half-wave, write y[m, r].
    const float bias2 = b2[0];
    #pragma unroll
    for (int j = 0; j < 8; ++j) {
        float s = yacc[j];
        s += __shfl_xor(s, 1, 32);
        s += __shfl_xor(s, 2, 32);
        s += __shfl_xor(s, 4, 32);
        s += __shfl_xor(s, 8, 32);
        if ((lane & 15) == 0) {
            int m = wave * 16 + j + ((lane >> 4) << 3);   // local row = batch index
            out[(size_t)m * NRES + r] = s + bias2;
        }
    }
}

extern "C" void kernel_launch(void* const* d_in, const int* in_sizes, int n_in,
                              void* d_out, int out_size, void* d_ws, size_t ws_size,
                              hipStream_t stream) {
    const float* s_s = (const float*)d_in[0];
    const float* W1  = (const float*)d_in[1];
    const float* b1  = (const float*)d_in[2];
    const float* W2  = (const float*)d_in[3];
    const float* b2  = (const float*)d_in[4];
    float* out = (float*)d_out;
    unsigned short* wb = (unsigned short*)d_ws;   // 2 MB packed bf16 W1

    // Pack W1 once per launch (stream-ordered before the GEMM).
    pack_w1_bf16<<<(DIN * HID) / 256, 256, 0, stream>>>(W1, wb);

    // 260 KB dynamic LDS per block (WGP has 320 KB).
    const size_t lds_bytes = (size_t)BATCH * LDS_STRIDE * sizeof(unsigned short);
    (void)hipFuncSetAttribute((const void*)fused_mlp,
                              hipFuncAttributeMaxDynamicSharedMemorySize,
                              (int)lds_bytes);
    fused_mlp<<<NRES, 256, lds_bytes, stream>>>(s_s, wb, b1, W2, b2, out);
}
